// EuclideanCodebook_56289841382016
// MI455X (gfx1250) — matrix operand order repaired and checked by reading, not compiled, not run on hardware
//
#include <hip/hip_runtime.h>
#include <stdint.h>

#define DECAY 0.8f
#define OMD   0.2f
#define EPS   1e-5f
#define K_CODES 2048
#define DIM     256
#define N_ROWS  32768  // 16*2048

typedef __bf16 bf16;
typedef bf16         v16bf __attribute__((ext_vector_type(16)));
typedef float        v8f   __attribute__((ext_vector_type(8)));
typedef unsigned int u32x4 __attribute__((ext_vector_type(4)));
typedef int          v4i   __attribute__((ext_vector_type(4)));

typedef __attribute__((address_space(1))) v4i* gptr_v4i;   // global
typedef __attribute__((address_space(3))) v4i* lptr_v4i;   // LDS

union FragAB { v16bf v; u32x4 u[2]; };

// ---- CDNA5 async global->LDS copy (ASYNCcnt path), with safe fallback ------
__device__ __forceinline__ void async_cp16(const bf16* src, bf16* dst) {
#if __has_builtin(__builtin_amdgcn_global_load_async_to_lds_b128)
    __builtin_amdgcn_global_load_async_to_lds_b128(
        (gptr_v4i)src, (lptr_v4i)dst, 0, 0);
#else
    *(u32x4*)dst = *(const u32x4*)src;   // synchronous fallback
#endif
}

__device__ __forceinline__ void wait_async_all() {
#if __has_builtin(__builtin_amdgcn_s_wait_asynccnt)
    __builtin_amdgcn_s_wait_asynccnt(0);
#else
    asm volatile("s_wait_asynccnt 0" ::: "memory");
#endif
}

// ---------------------------------------------------------------- utilities
__global__ __launch_bounds__(256) void k_zero(float* p, int n) {
    int i = blockIdx.x * 256 + threadIdx.x;
    if (i < n) p[i] = 0.0f;
}

__global__ __launch_bounds__(256) void k_cvt_x(const float* __restrict__ x,
                                               bf16* __restrict__ xbf, int n) {
    int i = blockIdx.x * 256 + threadIdx.x;
    if (i < n) xbf[i] = (bf16)x[i];
}

// one block per code: bf16 copy of the row + ||e||^2 in f32
__global__ __launch_bounds__(256) void k_prep_embed(const float* __restrict__ embed,
                                                    bf16* __restrict__ ebf,
                                                    float* __restrict__ norms) {
    __shared__ float red[256];
    int k = blockIdx.x, d = threadIdx.x;
    float e = embed[k * DIM + d];
    ebf[k * DIM + d] = (bf16)e;
    red[d] = e * e;
    __syncthreads();
    for (int s = 128; s > 0; s >>= 1) {
        if (d < s) red[d] += red[d + s];
        __syncthreads();
    }
    if (d == 0) norms[k] = red[0];
}

// ------------------------------------------------ score GEMM + argmax (bf16 WMMA)
// 128 threads = 4 waves; each wave: one 16-row M tile, full K=256 A slab in regs.
// Codebook double-buffered through LDS in 64-code tiles, prefetched with
// async global->LDS copies so the fetch of tile i+1 overlaps WMMAs on tile i.
__global__ __launch_bounds__(128)
void k_gemm_argmax(const bf16* __restrict__ xbf, const bf16* __restrict__ ebf,
                   const float* __restrict__ norms, int* __restrict__ ind,
                   float* __restrict__ out_ind) {
    // 264 = 256 + 8 bf16 pad (16B) -> conflict-free ds_load_b128; 2 x 33 KB buffers
    __shared__ __align__(16) bf16 Btile[2][64 * 264];

    const int tid  = threadIdx.x;
    const int wave = tid >> 5;
    const int lane = tid & 31;
    const int half = lane >> 4;
    const int r    = lane & 15;
    const int m0   = blockIdx.x * 64 + wave * 16;

    // this thread's slice of a 64x256 codebook tile: half a row (128 bf16)
    const int frow = tid >> 1;
    const int fcol = (tid & 1) * 128;

    // A fragments: lane holds row (lane&15); per frag two contiguous 16B chunks.
    FragAB A[8];
    const bf16* arow = xbf + (size_t)(m0 + r) * DIM;
#pragma unroll
    for (int kk = 0; kk < 8; ++kk) {
        int kb = kk * 32;
        A[kk].u[0] = *(const u32x4*)(arow + kb + 8 * half);
        A[kk].u[1] = *(const u32x4*)(arow + kb + 16 + 8 * half);
    }

    float bestv[8];
    int   besti[8];
#pragma unroll
    for (int j = 0; j < 8; ++j) { bestv[j] = -3.4e38f; besti[j] = 0; }

    // prologue: prefetch tile 0 into buffer 0
    {
        const bf16* src = ebf + (size_t)frow * DIM + fcol;
        bf16*       dst = &Btile[0][frow * 264 + fcol];
#pragma unroll
        for (int j = 0; j < 16; ++j) async_cp16(src + j * 8, dst + j * 8);
    }
    wait_async_all();
    __syncthreads();

    for (int c0 = 0; c0 < K_CODES; c0 += 64) {
        const int buf = (c0 >> 6) & 1;

        // kick off async prefetch of the next tile into the other buffer
        if (c0 + 64 < K_CODES) {
            const bf16* src = ebf + (size_t)(c0 + 64 + frow) * DIM + fcol;
            bf16*       dst = &Btile[buf ^ 1][frow * 264 + fcol];
#pragma unroll
            for (int j = 0; j < 16; ++j) async_cp16(src + j * 8, dst + j * 8);
        }

        // compute on the current buffer
#pragma unroll
        for (int t = 0; t < 4; ++t) {
            v8f acc = {};
#pragma unroll
            for (int kk = 0; kk < 8; ++kk) {
                // B fragment: lane holds column (lane&15); lanes 0-15 K 0..7,16..23; 16-31 K 8..15,24..31
                FragAB B;
                const bf16* brow = &Btile[buf][(t * 16 + r) * 264 + kk * 32 + 8 * half];
                B.u[0] = *(const u32x4*)(brow);
                B.u[1] = *(const u32x4*)(brow + 16);
                acc = __builtin_amdgcn_wmma_f32_16x16x32_bf16(
                    false, A[kk].v, false, B.v, (short)0, acc, false, false);
            }
            int   n  = c0 + t * 16 + r;     // this lane's column (code index)
            float nv = norms[n];
#pragma unroll
            for (int j = 0; j < 8; ++j) {
                float s = 2.0f * acc[j] - nv;  // -||x||^2 term is row-constant: argmax-invariant
                if (s > bestv[j]) { bestv[j] = s; besti[j] = n; }
            }
        }

        wait_async_all();   // my prefetch issues have landed in LDS
        __syncthreads();    // everyone's prefetch landed; everyone done reading old buffer
    }

    // reduce (max, argmax) across the 16 lanes of each half-wave (xor<=8 stays in half)
#pragma unroll
    for (int j = 0; j < 8; ++j) {
        float v = bestv[j]; int i = besti[j];
#pragma unroll
        for (int off = 8; off > 0; off >>= 1) {
            float ov = __shfl_xor(v, off, 32);
            int   oi = __shfl_xor(i, off, 32);
            if (ov > v || (ov == v && oi < i)) { v = ov; i = oi; }
        }
        bestv[j] = v; besti[j] = i;
    }
    if (r == 0) {
#pragma unroll
        for (int j = 0; j < 8; ++j) {
            int row = m0 + half * 8 + j;    // C layout: vgpr j -> M=j (+8 for upper half)
            ind[row]     = besti[j];
            out_ind[row] = (float)besti[j];
        }
    }
}

// ------------------------------------------------ gather + segment sums (atomics)
__global__ __launch_bounds__(256)
void k_scatter(const float* __restrict__ x, const float* __restrict__ embed,
               const int* __restrict__ ind, float* __restrict__ quant,
               float* __restrict__ esum, float* __restrict__ counts) {
    int row = blockIdx.x, d = threadIdx.x;
    int c = ind[row];
    quant[(size_t)row * DIM + d] = embed[c * DIM + d];
    atomicAdd(&esum[c * DIM + d], x[(size_t)row * DIM + d]);
    if (d == 0) atomicAdd(&counts[c], 1.0f);
}

// new_cluster_size + global sum n
__global__ __launch_bounds__(256)
void k_ncs(const float* __restrict__ cs, const float* __restrict__ counts,
           float* __restrict__ out_ncs, float* __restrict__ nsum) {
    __shared__ float red[256];
    int i = blockIdx.x * 256 + threadIdx.x;
    float v = cs[i] * DECAY + OMD * counts[i];
    out_ncs[i] = v;
    red[threadIdx.x] = v;
    __syncthreads();
    for (int s = 128; s > 0; s >>= 1) {
        if (threadIdx.x < s) red[threadIdx.x] += red[threadIdx.x + s];
        __syncthreads();
    }
    if (threadIdx.x == 0) atomicAdd(nsum, red[0]);
}

__global__ __launch_bounds__(256)
void k_finalize(const float* __restrict__ eavg, const float* __restrict__ esum,
                const float* __restrict__ ncs, const float* __restrict__ nsum,
                float* __restrict__ out_navg, float* __restrict__ out_nembed) {
    int k = blockIdx.x, d = threadIdx.x;
    float n        = *nsum;
    float smoothed = (ncs[k] + EPS) / (n + (float)K_CODES * EPS) * n;
    float nav      = eavg[k * DIM + d] * DECAY + OMD * esum[k * DIM + d];
    out_navg[k * DIM + d]   = nav;
    out_nembed[k * DIM + d] = nav / smoothed;
}

// ---------------------------------------------------------------- launcher
extern "C" void kernel_launch(void* const* d_in, const int* in_sizes, int n_in,
                              void* d_out, int out_size, void* d_ws, size_t ws_size,
                              hipStream_t stream) {
    const float* x     = (const float*)d_in[0];  // [16,2048,256]
    const float* embed = (const float*)d_in[1];  // [2048,256]
    const float* cs    = (const float*)d_in[2];  // [2048]
    const float* eavg  = (const float*)d_in[3];  // [2048,256]

    float* out        = (float*)d_out;
    float* out_quant  = out;                 // 8388608
    float* out_ind    = out + 8388608;       // 32768
    float* out_nembed = out + 8421376;       // 524288
    float* out_ncs    = out + 8945664;       // 2048
    float* out_navg   = out + 8947712;       // 524288

    char*  ws     = (char*)d_ws;             // needs ~20.1 MB total
    bf16*  xbf    = (bf16*) (ws);                     // 16 MB
    bf16*  ebf    = (bf16*) (ws + 16777216);          // 1 MB
    float* norms  = (float*)(ws + 17825792);          // 8 KB
    int*   ind    = (int*)  (ws + 17833984);          // 128 KB
    float* counts = (float*)(ws + 17965056);          // 8 KB  (zero region start)
    float* esum   = (float*)(ws + 17973248);          // 2 MB  (zero region)
    float* nsum   = (float*)(ws + 20070400);          // 4 B   (zero region end)

    // zero counts + esum + nsum (contiguous): 2048 + 524288 + 1 floats
    k_zero<<<2057, 256, 0, stream>>>(counts, 2048 + 524288 + 1);
    k_cvt_x<<<32768, 256, 0, stream>>>(x, xbf, N_ROWS * DIM);
    k_prep_embed<<<K_CODES, 256, 0, stream>>>(embed, ebf, norms);
    k_gemm_argmax<<<N_ROWS / 64, 128, 0, stream>>>(xbf, ebf, norms, ind, out_ind);
    k_scatter<<<N_ROWS, 256, 0, stream>>>(x, embed, ind, out_quant, esum, counts);
    k_ncs<<<K_CODES / 256, 256, 0, stream>>>(cs, counts, out_ncs, nsum);
    k_finalize<<<K_CODES, 256, 0, stream>>>(eavg, esum, out_ncs, nsum, out_navg, out_nembed);
}
